// GruAgent_9268539425220
// MI455X (gfx1250) — matrix operand order repaired and checked by reading, not compiled, not run on hardware
//
#include <hip/hip_runtime.h>

#define TT 512
#define BB 512
#define NROWS (TT * BB)   // 262144

typedef __attribute__((ext_vector_type(16))) __bf16 v16bf;
typedef __attribute__((ext_vector_type(8)))  float  v8f;

__device__ __forceinline__ v8f wmma_bf16(v16bf a, v16bf b, v8f c) {
    return __builtin_amdgcn_wmma_f32_16x16x32_bf16(false, a, false, b,
                                                   (short)0, c, false, false);
}

// ---- CDNA5 async global->LDS copy (ASYNCcnt), via inline asm --------------
// Syntax: global_load_async_to_lds_b128 vdst(LDS byte addr), vaddr(64b), off
__device__ __forceinline__ void async_b128(const void* g, void* l) {
    unsigned lo = (unsigned)(__UINTPTR_TYPE__)(__attribute__((address_space(3))) char*)(l);
    asm volatile("global_load_async_to_lds_b128 %0, %1, off"
                 :: "v"(lo), "v"(g) : "memory");
}

#if __has_builtin(__builtin_amdgcn_s_wait_asynccnt)
#define WAIT_ASYNC(n) __builtin_amdgcn_s_wait_asynccnt(n)
#else
#define WAIT_ASYNC(n) asm volatile("s_wait_asynccnt %0" :: "i"(n) : "memory")
#endif

// A-fragment: 16x32 bf16 tile of a row-major matrix (rows = M, cols = K).
// Lane L holds M = L&15; VGPR j holds K = kb + (j>>2)*16 + (L>>4)*8 + (j&3)*2 (pair).
template <typename T>
__device__ __forceinline__ v16bf load_a_frag(const T* tile, int stride, int kb, int lane) {
    int m = lane & 15, hi = lane >> 4;
    const T* r = tile + (size_t)m * stride;
    v16bf f;
#pragma unroll
    for (int j = 0; j < 8; ++j) {
        int k = kb + ((j >> 2) << 4) + (hi << 3) + ((j & 3) << 1);
        f[2 * j]     = (__bf16)(float)r[k];
        f[2 * j + 1] = (__bf16)(float)r[k + 1];
    }
    return f;
}

// B-fragment: 32x16 bf16 tile where B[k,n] = w[n0+n, k] (w row-major [N,K]).
// Lane L holds N = L&15; VGPR j holds K = kb + (j>>2)*16 + (L>>4)*8 + (j&3)*2 (pair).
template <typename T>
__device__ __forceinline__ v16bf load_b_frag(const T* w, int n0, int stride, int kb,
                                             int lane, int nmax) {
    int n  = (lane & 15) + n0;
    int hi = lane >> 4;
    bool ok = n < nmax;
    const T* r = w + (size_t)(ok ? n : 0) * stride;
    v16bf f;
#pragma unroll
    for (int j = 0; j < 8; ++j) {
        int k = kb + ((j >> 2) << 4) + (hi << 3) + ((j & 3) << 1);
        f[2 * j]     = (__bf16)(ok ? (float)r[k]     : 0.0f);
        f[2 * j + 1] = (__bf16)(ok ? (float)r[k + 1] : 0.0f);
    }
    return f;
}

__device__ __forceinline__ float sigmoidf(float x) { return 1.0f / (1.0f + __expf(-x)); }

// bf16 weight arena element offsets (within ws)
#define OFF_WIH 0
#define OFF_WHH 12288
#define OFF_AW1 24576
#define OFF_AW2 32768
#define OFF_AW3 36864
#define OFF_CW1 37248
#define OFF_CW2 45440
#define OFF_CW3 49536
#define W_TOTAL 49600

// ---------------------------------------------------------------------------
// Phase 0: one-time f32 -> bf16 conversion of all weight matrices.
// ---------------------------------------------------------------------------
__global__ void __launch_bounds__(256) cvt_weights(const float* __restrict__ w_ih,
                                                   const float* __restrict__ w_hh,
                                                   const float* __restrict__ aw1,
                                                   const float* __restrict__ aw2,
                                                   const float* __restrict__ aw3,
                                                   const float* __restrict__ cw1,
                                                   const float* __restrict__ cw2,
                                                   const float* __restrict__ cw3,
                                                   __bf16* __restrict__ wb) {
    int i = blockIdx.x * 256 + threadIdx.x;
    if (i >= W_TOTAL) return;
    float v;
    if      (i < OFF_WHH) v = w_ih[i - OFF_WIH];
    else if (i < OFF_AW1) v = w_hh[i - OFF_WHH];
    else if (i < OFF_AW2) v = aw1[i - OFF_AW1];
    else if (i < OFF_AW3) v = aw2[i - OFF_AW2];
    else if (i < OFF_CW1) v = aw3[i - OFF_AW3];
    else if (i < OFF_CW2) v = cw1[i - OFF_CW1];
    else if (i < OFF_CW3) v = cw2[i - OFF_CW2];
    else                  v = cw3[i - OFF_CW3];
    wb[i] = (__bf16)v;
}

// ---------------------------------------------------------------------------
// Phase 1: gi = x @ w_ih^T + b_ih  ->  bf16 [262144,192]
// One 16-row M-tile per wave; 12 N-tiles x 2 K-halves = 24 WMMAs per wave.
// ---------------------------------------------------------------------------
__global__ void __launch_bounds__(128) gi_kernel(const float*  __restrict__ x,
                                                 const __bf16* __restrict__ w_ih,
                                                 const float*  __restrict__ b_ih,
                                                 __bf16* __restrict__ gi) {
    int lane = threadIdx.x & 31, wv = threadIdx.x >> 5;
    size_t row0 = ((size_t)blockIdx.x * 4 + wv) * 16;
    const float* xt = x + row0 * 64;

    v16bf a0 = load_a_frag(xt, 64, 0, lane);
    v16bf a1 = load_a_frag(xt, 64, 32, lane);

    int nn = lane & 15, hi = lane >> 4;
#pragma unroll
    for (int nt = 0; nt < 12; ++nt) {
        v16bf b0 = load_b_frag(w_ih, nt * 16, 64, 0, lane, 192);
        v16bf b1 = load_b_frag(w_ih, nt * 16, 64, 32, lane, 192);
        v8f c = {};
        c = wmma_bf16(a0, b0, c);
        c = wmma_bf16(a1, b1, c);
        float bias = b_ih[nt * 16 + nn];
#pragma unroll
        for (int v = 0; v < 8; ++v)
            gi[(row0 + v + hi * 8) * 192 + nt * 16 + nn] = (__bf16)(c[v] + bias);
    }
}

// ---------------------------------------------------------------------------
// Phase 2: sequential GRU scan. 32 blocks, each owns a 16-row batch slice for
// all 512 steps. h in LDS, w_hh fragments persistent in VGPRs, and the next
// step's gi tile (6 KB) double-buffered into LDS via async global->LDS copies
// so its latency overlaps the current step's WMMA + gate math.
// ---------------------------------------------------------------------------
__global__ void __launch_bounds__(128) gru_scan_kernel(const float*  __restrict__ done,
                                                       const float*  __restrict__ gru_state,
                                                       const __bf16* __restrict__ w_hh,
                                                       const float*  __restrict__ b_hh,
                                                       const __bf16* __restrict__ gi,
                                                       __bf16*       __restrict__ hidden) {
    __shared__ float h[16 * 64];
    __shared__ float gh[16 * 192];
    __shared__ __align__(16) __bf16 gib[2][16 * 192];

    int tid = threadIdx.x, lane = tid & 31, wv = tid >> 5;
    int b0 = blockIdx.x * 16;
    int nn = lane & 15, hi = lane >> 4;

    // init h from gru_state [1,B,H]
    for (int idx = tid; idx < 16 * 64; idx += 128)
        h[idx] = gru_state[(size_t)(b0 + (idx >> 6)) * 64 + (idx & 63)];

    // persistent w_hh fragments: wave wv covers gate-output N-tiles wv*3 .. wv*3+2
    v16bf bf[3][2];
#pragma unroll
    for (int t3 = 0; t3 < 3; ++t3)
#pragma unroll
        for (int kh = 0; kh < 2; ++kh)
            bf[t3][kh] = load_b_frag(w_hh, (wv * 3 + t3) * 16, 64, kh * 32, lane, 192);

    // kick off async copy of gi tile for t = 0 (6144 B = 128 thr x 3 x 16 B)
    {
        const char* g = (const char*)(gi + (size_t)b0 * 192);
        char* l = (char*)gib[0];
#pragma unroll
        for (int i = 0; i < 3; ++i)
            async_b128(g + (tid + 128 * i) * 16, l + (tid + 128 * i) * 16);
    }
    __syncthreads();

    for (int t = 0; t < TT; ++t) {
        size_t rowbase = (size_t)t * BB + b0;

        // prefetch next step's gi tile into the other LDS buffer
        if (t + 1 < TT) {
            const char* g = (const char*)(gi + (rowbase + BB) * 192);
            char* l = (char*)gib[(t + 1) & 1];
#pragma unroll
            for (int i = 0; i < 3; ++i)
                async_b128(g + (tid + 128 * i) * 16, l + (tid + 128 * i) * 16);
        }

        // reset-mask h on done (reference: h = (1-done)*h before the step)
        for (int idx = tid; idx < 16 * 64; idx += 128)
            h[idx] *= (1.0f - done[rowbase + (idx >> 6)]);
        __syncthreads();

        // gh = h @ w_hh^T for this wave's 3 N-tiles
        v16bf a0 = load_a_frag(h, 64, 0, lane);
        v16bf a1 = load_a_frag(h, 64, 32, lane);
#pragma unroll
        for (int t3 = 0; t3 < 3; ++t3) {
            v8f c = {};
            c = wmma_bf16(a0, bf[t3][0], c);
            c = wmma_bf16(a1, bf[t3][1], c);
            int n0 = (wv * 3 + t3) * 16;
#pragma unroll
            for (int v = 0; v < 8; ++v)
                gh[(v + hi * 8) * 192 + n0 + nn] = c[v];
        }

        // this wave's copies for step t (issued last iteration) are the oldest:
        // allow the 3 just-issued (t+1) to remain outstanding.
        if (t + 1 < TT) { WAIT_ASYNC(3); } else { WAIT_ASYNC(0); }
        __syncthreads();

        // elementwise gate math + state update + hidden output
        const __bf16* gt = gib[t & 1];
        for (int idx = tid; idx < 16 * 64; idx += 128) {
            int m = idx >> 6, j = idx & 63;
            float ir = (float)gt[m * 192 + j];
            float iz = (float)gt[m * 192 + 64 + j];
            float in = (float)gt[m * 192 + 128 + j];
            float hr = gh[m * 192 + j]        + b_hh[j];
            float hz = gh[m * 192 + 64 + j]   + b_hh[64 + j];
            float hn = gh[m * 192 + 128 + j]  + b_hh[128 + j];
            float r = sigmoidf(ir + hr);
            float z = sigmoidf(iz + hz);
            float n = tanhf(in + r * hn);
            float hnew = (1.0f - z) * n + z * h[idx];
            h[idx] = hnew;
            hidden[(rowbase + m) * 64 + j] = (__bf16)hnew;
        }
        __syncthreads();
    }
}

// ---------------------------------------------------------------------------
// Phase 3: fused actor + critic MLPs on cat = [hidden(bf16), x] (K=128).
// One 16-row tile per wave; layer outputs transposed via per-wave LDS buffer
// (in-order LDS pipe makes store->frag-load safe within a wave).
// ---------------------------------------------------------------------------
__global__ void __launch_bounds__(128) mlp_kernel(const __bf16* __restrict__ hidden,
                                                  const float*  __restrict__ x,
                                                  const __bf16* __restrict__ wb,
                                                  const float* __restrict__ ab1,
                                                  const float* __restrict__ ab2,
                                                  const float* __restrict__ ab3,
                                                  const float* __restrict__ cb1,
                                                  const float* __restrict__ cb2,
                                                  const float* __restrict__ cb3,
                                                  float* __restrict__ out) {
    __shared__ float buf[4][16 * 64];
    int lane = threadIdx.x & 31, wv = threadIdx.x >> 5;
    size_t row0 = ((size_t)blockIdx.x * 4 + wv) * 16;
    float* my = buf[wv];
    int nn = lane & 15, hi = lane >> 4;

    // cat A-fragments: K 0..63 from hidden (bf16), K 64..127 from x (f32)
    v16bf af[4];
    af[0] = load_a_frag(hidden + row0 * 64, 64, 0, lane);
    af[1] = load_a_frag(hidden + row0 * 64, 64, 32, lane);
    af[2] = load_a_frag(x + row0 * 64, 64, 0, lane);
    af[3] = load_a_frag(x + row0 * 64, 64, 32, lane);

    for (int head = 0; head < 2; ++head) {
        const __bf16* w1 = wb + (head ? OFF_CW1 : OFF_AW1);
        const __bf16* w2 = wb + (head ? OFF_CW2 : OFF_AW2);
        const __bf16* w3 = wb + (head ? OFF_CW3 : OFF_AW3);
        const float* bb1 = head ? cb1 : ab1;
        const float* bb2 = head ? cb2 : ab2;
        const float* bb3 = head ? cb3 : ab3;
        int n3 = head ? 1 : 6;

        // layer 1: [16,128] @ [128->64], tanh
#pragma unroll
        for (int nt = 0; nt < 4; ++nt) {
            v8f c = {};
#pragma unroll
            for (int kf = 0; kf < 4; ++kf) {
                v16bf bq = load_b_frag(w1, nt * 16, 128, kf * 32, lane, 64);
                c = wmma_bf16(af[kf], bq, c);
            }
            float bias = bb1[nt * 16 + nn];
#pragma unroll
            for (int v = 0; v < 8; ++v)
                my[(v + hi * 8) * 64 + nt * 16 + nn] = tanhf(c[v] + bias);
        }

        // layer 2: [16,64] @ [64->64], tanh (frag loads precede the stores)
        v16bf a20 = load_a_frag(my, 64, 0, lane);
        v16bf a21 = load_a_frag(my, 64, 32, lane);
#pragma unroll
        for (int nt = 0; nt < 4; ++nt) {
            v16bf bq0 = load_b_frag(w2, nt * 16, 64, 0, lane, 64);
            v16bf bq1 = load_b_frag(w2, nt * 16, 64, 32, lane, 64);
            v8f c = {};
            c = wmma_bf16(a20, bq0, c);
            c = wmma_bf16(a21, bq1, c);
            float bias = bb2[nt * 16 + nn];
#pragma unroll
            for (int v = 0; v < 8; ++v)
                my[(v + hi * 8) * 64 + nt * 16 + nn] = tanhf(c[v] + bias);
        }

        // layer 3: [16,64] @ [64 -> n3] (single N-tile, zero-padded rows)
        v16bf a30 = load_a_frag(my, 64, 0, lane);
        v16bf a31 = load_a_frag(my, 64, 32, lane);
        v16bf bq0 = load_b_frag(w3, 0, 64, 0, lane, n3);
        v16bf bq1 = load_b_frag(w3, 0, 64, 32, lane, n3);
        v8f c = {};
        c = wmma_bf16(a30, bq0, c);
        c = wmma_bf16(a31, bq1, c);

        if (head == 0) {
            if (nn < 6) {
                float bias = bb3[nn];
#pragma unroll
                for (int v = 0; v < 8; ++v)
                    out[(row0 + v + hi * 8) * 7 + nn] = c[v] + bias;
            }
        } else {
            if (nn == 0) {
                float bias = bb3[0];
#pragma unroll
                for (int v = 0; v < 8; ++v)
                    out[(row0 + v + hi * 8) * 7 + 6] = c[v] + bias;
            }
        }
    }
}

extern "C" void kernel_launch(void* const* d_in, const int* in_sizes, int n_in,
                              void* d_out, int out_size, void* d_ws, size_t ws_size,
                              hipStream_t stream) {
    const float* x         = (const float*)d_in[0];
    const float* done      = (const float*)d_in[1];
    const float* gru_state = (const float*)d_in[2];
    const float* w_ih      = (const float*)d_in[3];
    const float* w_hh      = (const float*)d_in[4];
    const float* b_ih      = (const float*)d_in[5];
    const float* b_hh      = (const float*)d_in[6];
    const float* aw1 = (const float*)d_in[7];  const float* ab1 = (const float*)d_in[8];
    const float* aw2 = (const float*)d_in[9];  const float* ab2 = (const float*)d_in[10];
    const float* aw3 = (const float*)d_in[11]; const float* ab3 = (const float*)d_in[12];
    const float* cw1 = (const float*)d_in[13]; const float* cb1 = (const float*)d_in[14];
    const float* cw2 = (const float*)d_in[15]; const float* cb2 = (const float*)d_in[16];
    const float* cw3 = (const float*)d_in[17]; const float* cb3 = (const float*)d_in[18];
    float* out = (float*)d_out;

    // workspace: gi bf16 [NROWS,192] | hidden bf16 [NROWS,64] | bf16 weights
    __bf16* gi     = (__bf16*)d_ws;
    __bf16* hidden = gi + (size_t)NROWS * 192;
    __bf16* wb     = hidden + (size_t)NROWS * 64;

    cvt_weights<<<(W_TOTAL + 255) / 256, 256, 0, stream>>>(w_ih, w_hh, aw1, aw2, aw3,
                                                           cw1, cw2, cw3, wb);
    gi_kernel<<<NROWS / 64, 128, 0, stream>>>(x, wb + OFF_WIH, b_ih, gi);
    gru_scan_kernel<<<BB / 16, 128, 0, stream>>>(done, gru_state, wb + OFF_WHH, b_hh,
                                                 gi, hidden);
    mlp_kernel<<<NROWS / 64, 128, 0, stream>>>(hidden, x, wb,
                                               ab1, ab2, ab3, cb1, cb2, cb3, out);
}
